// BidirReGULayer_32667521253543
// MI455X (gfx1250) — matrix operand order repaired and checked by reading, not compiled, not run on hardware
//
#include <hip/hip_runtime.h>
#include <cstdint>

// ---------------------------------------------------------------------------
// Bidirectional ReGU layer for MI455X (gfx1250).
//   Phase 1: convert x / weights to bf16 (workspace).
//   Phase 2: parallel WMMA GEMM Pre = x @ [Wih;Wc]^T (+bias), f32 out.
//            A-slab staged through LDS with double-buffered
//            global_load_async_to_lds_b128 (ASYNCcnt) copies.
//   Phase 3: persistent recurrent kernel, 1 workgroup per direction,
//            h kept in LDS as bf16 (swizzled), gates via v_wmma_f32_16x16x32_bf16,
//            next-step Pre/x slabs prefetched (global_prefetch_b8).
// Workspace: ~442 MB (Xbf 32MB, weights 5MB, Pre 384MB f32, c-state 0.25MB).
// ---------------------------------------------------------------------------

#define T_STEPS 512
#define BATCH   64
#define IDIM    512
#define HDIM    512
#define H2      1024
#define H3      1536
#define TB      (T_STEPS * BATCH)   // 32768

typedef unsigned short u16;
typedef __attribute__((ext_vector_type(16))) __bf16 v16bf;
typedef __attribute__((ext_vector_type(8)))  float  v8f;

union FragBF { uint4 q[2]; v16bf v; };   // 32 bytes: one 16x32/32x16 bf16 fragment

__device__ __forceinline__ u16 f2bf(float f) {
  union { float f; unsigned int u; } c; c.f = f;
  unsigned int u = c.u;
  if ((u & 0x7fffffffu) > 0x7f800000u) return (u16)((u >> 16) | 0x0040u); // quiet NaN
  unsigned int r = 0x7fffu + ((u >> 16) & 1u);                            // RNE
  return (u16)((u + r) >> 16);
}

__device__ __forceinline__ float sigmoidf_(float x) {
  return 1.0f / (1.0f + __expf(-x));
}

__device__ __forceinline__ float tanhf_(float x) {
#if __has_builtin(__builtin_amdgcn_tanhf)
  return __builtin_amdgcn_tanhf(x);
#else
  return tanhf(x);
#endif
}

__device__ __forceinline__ v8f vzero8() {
  v8f z;
#pragma unroll
  for (int i = 0; i < 8; ++i) z[i] = 0.0f;
  return z;
}

// ---------------------------------------------------------------------------
// f32 -> bf16 conversion
// ---------------------------------------------------------------------------
__global__ void cvt_bf16_kernel(const float* __restrict__ src,
                                u16* __restrict__ dst, int n) {
  int i = blockIdx.x * 256 + threadIdx.x;
  if (i < n) dst[i] = f2bf(src[i]);
}

// ---------------------------------------------------------------------------
// Pre[d][m][n] = sum_k Xbf[m][k] * Wcat[d][n][k]  (+ bias for n < 2H)
//   m in [0,TB), n in [0,H3). 256 threads = 8 waves; each wave: 16 rows x 64 cols.
//   A slab (128 rows x 32 k, 8 KB bf16) staged in LDS via async global->LDS
//   copies, double-buffered against the WMMA pipeline.
// ---------------------------------------------------------------------------
#define A_STRIDE 40   // u16 per row (32 data + 8 pad -> 80B stride, conflict-free)

__global__ __launch_bounds__(256, 2)
void pregemm_kernel(const u16* __restrict__ Xbf,      // [TB][IDIM]
                    const u16* __restrict__ Wcat,     // [2][H3][IDIM]
                    const float* __restrict__ bF,
                    const float* __restrict__ bB,
                    float* __restrict__ Pre) {        // [2][TB][H3]
  __shared__ u16 aS[2][128 * A_STRIDE];               // 2 x 10 KB

  const int tid  = threadIdx.x;
  const int d    = blockIdx.z;
  const int lane = tid & 31;
  const int wv   = tid >> 5;
  const int hi   = lane >> 4;     // lane half: selects K block in fragments
  const int l15  = lane & 15;
  const int mbase = blockIdx.x * 128;       // 8 M-tiles per block
  const int nb    = blockIdx.y * 64;        // 64-col N slab (4 tiles)

  const u16* W     = Wcat + (size_t)d * H3 * IDIM;
  const float* bia = d ? bB : bF;
  float* P         = Pre + (size_t)d * TB * H3;

  // Async-stage one 128x32 A k-slab into LDS buffer `buf`.
  // 512 16B chunks; each thread issues 2 -> 2 async instructions per wave.
  auto issueA = [&](int buf, int kk) {
#pragma unroll
    for (int p = 0; p < 2; ++p) {
      const int i   = tid + p * 256;
      const int row = i >> 2;               // 0..127
      const int ch  = i & 3;                // 16B chunk within row
      const u16* g  = Xbf + (size_t)(mbase + row) * IDIM + kk + ch * 8;
      const uint32_t lds = (uint32_t)(uintptr_t)&aS[buf][row * A_STRIDE + ch * 8];
      asm volatile("global_load_async_to_lds_b128 %0, %1, off"
                   :: "v"(lds), "v"(g) : "memory");
    }
  };

  v8f acc[4];
#pragma unroll
  for (int nt = 0; nt < 4; ++nt) acc[nt] = vzero8();

  issueA(0, 0);
  int cur = 0;
  for (int kk = 0; kk < IDIM; kk += 32) {
    const bool more = (kk + 32) < IDIM;
    if (more) {
      issueA(cur ^ 1, kk + 32);
      asm volatile("s_wait_asynccnt 0x2" ::: "memory");  // cur slab landed
    } else {
      asm volatile("s_wait_asynccnt 0x0" ::: "memory");
    }
    __syncthreads();

    // A fragment from LDS: element e -> k = hi*16+e
    FragBF a;
    const u16* ar = &aS[cur][(wv * 16 + l15) * A_STRIDE];
    a.q[0] = *(const uint4*)(ar + hi * 16);
    a.q[1] = *(const uint4*)(ar + hi * 16 + 8);
#pragma unroll
    for (int nt = 0; nt < 4; ++nt) {
      // B fragment: element e -> B[kk + hi*16 + e][n], B[k][n] = W[n][k]
      const u16* br = W + (size_t)(nb + nt * 16 + l15) * IDIM + kk + hi * 16;
      FragBF b;
      b.q[0] = *(const uint4*)(br);
      b.q[1] = *(const uint4*)(br + 8);
      acc[nt] = __builtin_amdgcn_wmma_f32_16x16x32_bf16(
          false, a.v, false, b.v, (short)0, acc[nt], false, false);
    }
    __syncthreads();   // done reading `cur` before it is refilled
    cur ^= 1;
  }

#pragma unroll
  for (int nt = 0; nt < 4; ++nt) {
    const int n = nb + nt * 16 + l15;
    const float bb = (n < H2) ? bia[n] : 0.0f;
#pragma unroll
    for (int r = 0; r < 8; ++r) {
      const int m = mbase + wv * 16 + hi * 8 + r;   // C/D layout: VGPR r
      P[(size_t)m * H3 + n] = acc[nt][r] + bb;
    }
  }
}

// ---------------------------------------------------------------------------
// Persistent recurrent scan: one workgroup per direction (blockIdx.x = 0 fwd, 1 bwd).
// 512 threads = 16 waves. Wave w owns hidden columns {w*16.., (w+16)*16..} (2 groups).
// h state in LDS as bf16, 8-element chunks XOR-swizzled by row to kill bank conflicts.
// c state in f32 workspace (each element touched by exactly one lane).
// ---------------------------------------------------------------------------
__global__ __launch_bounds__(512, 1)
void regu_recurrent_kernel(const float* __restrict__ X,     // [T][B][IDIM] f32
                           const u16* __restrict__ WhhB,    // [2][H2][HDIM] bf16
                           const float* __restrict__ Pre,   // [2][TB][H3] f32
                           const float* __restrict__ h0F, const float* __restrict__ c0F,
                           const float* __restrict__ h0B, const float* __restrict__ c0B,
                           float* __restrict__ cWS,         // [2][B][HDIM] f32
                           float* __restrict__ out) {       // outputs ++ states
  __shared__ uint4 hS4[BATCH * (HDIM / 8)];   // 64 KB: 64 rows x 64 chunks (8 bf16)
  u16* hS = (u16*)hS4;

  const int d    = blockIdx.x;
  const int tid  = threadIdx.x;
  const int lane = tid & 31;
  const int w    = tid >> 5;       // wave 0..15
  const int hi   = lane >> 4;
  const int l15  = lane & 15;

  const u16*  Whh = WhhB + (size_t)d * H2 * HDIM;
  const float* P  = Pre + (size_t)d * TB * H3;
  const float* h0 = d ? h0B : h0F;
  const float* c0 = d ? c0B : c0F;
  float* cS       = cWS + (size_t)d * BATCH * HDIM;

  // init h (bf16, swizzled) and c
  for (int i = tid; i < BATCH * HDIM; i += 512) {
    const int b = i >> 9, j = i & (HDIM - 1);
    hS[(b << 9) + (((j >> 3) ^ (b & 15)) << 3) + (j & 7)] = f2bf(h0[i]);
    cS[i] = c0[i];
  }
  __syncthreads();

  v8f accF[2][4], accO[2][4];

  for (int s = 0; s < T_STEPS; ++s) {
    const int t = d ? (T_STEPS - 1 - s) : s;

    // prefetch next step's Pre / x slabs while this step's GEMM runs
    if (s + 1 < T_STEPS) {
      const int t2 = d ? (T_STEPS - 2 - s) : (s + 1);
      const float* pn = P + (size_t)t2 * BATCH * H3;
      const float* xn = X + (size_t)t2 * BATCH * IDIM;
      __builtin_prefetch(pn + (size_t)tid * 192, 0, 1);
      __builtin_prefetch(pn + (size_t)tid * 192 + 96, 0, 1);
      __builtin_prefetch(xn + (size_t)tid * 64, 0, 1);
    }

#pragma unroll
    for (int jgi = 0; jgi < 2; ++jgi)
#pragma unroll
      for (int mt = 0; mt < 4; ++mt) { accF[jgi][mt] = vzero8(); accO[jgi][mt] = vzero8(); }

    // gates_acc = h @ Whh^T   (A = h[64x512] bf16 in LDS, B = Whh rows, K contiguous)
    for (int kk = 0; kk < HDIM; kk += 32) {
      FragBF a[4];
#pragma unroll
      for (int mt = 0; mt < 4; ++mt) {
        const int m   = mt * 16 + l15;
        const int ch0 = (kk >> 3) + hi * 2;      // logical 8-elem chunk index
        a[mt].q[0] = hS4[m * 64 + ((ch0    ) ^ (m & 15))];
        a[mt].q[1] = hS4[m * 64 + ((ch0 + 1) ^ (m & 15))];
      }
#pragma unroll
      for (int jgi = 0; jgi < 2; ++jgi) {
        const int jg = w + jgi * 16;             // column group 0..31
        const u16* rF = Whh + (size_t)(jg * 16 + l15) * HDIM + kk + hi * 16;
        const u16* rO = Whh + (size_t)(HDIM + jg * 16 + l15) * HDIM + kk + hi * 16;
        FragBF bf_, bo_;
        bf_.q[0] = *(const uint4*)(rF); bf_.q[1] = *(const uint4*)(rF + 8);
        bo_.q[0] = *(const uint4*)(rO); bo_.q[1] = *(const uint4*)(rO + 8);
#pragma unroll
        for (int mt = 0; mt < 4; ++mt) {
          accF[jgi][mt] = __builtin_amdgcn_wmma_f32_16x16x32_bf16(
              false, a[mt].v, false, bf_.v, (short)0, accF[jgi][mt], false, false);
          accO[jgi][mt] = __builtin_amdgcn_wmma_f32_16x16x32_bf16(
              false, a[mt].v, false, bo_.v, (short)0, accO[jgi][mt], false, false);
        }
      }
    }
    __syncthreads();   // all LDS h reads done before overwrite

    const float* prow = P + (size_t)t * BATCH * H3;
    const float* xrow = X + (size_t)t * BATCH * IDIM;
    float* orow = out + (size_t)t * BATCH * H2 + d * HDIM;

#pragma unroll
    for (int jgi = 0; jgi < 2; ++jgi) {
      const int j = (w + jgi * 16) * 16 + l15;   // hidden column 0..511
#pragma unroll
      for (int mt = 0; mt < 4; ++mt) {
#pragma unroll
        for (int r = 0; r < 8; ++r) {
          const int m = mt * 16 + hi * 8 + r;    // batch row (C/D layout)
          const size_t pb = (size_t)m * H3;
          const float f = sigmoidf_(accF[jgi][mt][r] + prow[pb + j]);
          const float o = sigmoidf_(accO[jgi][mt][r] + prow[pb + HDIM + j]);
          const float cOld = cS[m * HDIM + j];
          const float cNew = f * cOld + (1.0f - f) * tanhf_(prow[pb + H2 + j]);
          cS[m * HDIM + j] = cNew;
          const float x = xrow[(size_t)m * IDIM + j];
          const float h = o * tanhf_(cNew) + (1.0f - o) * x;
          orow[(size_t)m * H2 + j] = h;
          hS[(m << 9) + (((j >> 3) ^ (m & 15)) << 3) + (j & 7)] = f2bf(h);
          if (s == T_STEPS - 1) {                // final carry for this direction
            float* st = out + (size_t)TB * H2 + (size_t)d * 2 * BATCH * HDIM;
            st[m * HDIM + j] = h;                // h state
            st[BATCH * HDIM + m * HDIM + j] = cNew;  // c state
          }
        }
      }
    }
    __syncthreads();
  }
}

// ---------------------------------------------------------------------------
extern "C" void kernel_launch(void* const* d_in, const int* in_sizes, int n_in,
                              void* d_out, int out_size, void* d_ws, size_t ws_size,
                              hipStream_t stream) {
  const float* X    = (const float*)d_in[0];
  const float* h0F  = (const float*)d_in[1];
  const float* c0F  = (const float*)d_in[2];
  const float* h0B  = (const float*)d_in[3];
  const float* c0B  = (const float*)d_in[4];
  const float* WihF = (const float*)d_in[5];
  const float* WhhF = (const float*)d_in[6];
  const float* bF   = (const float*)d_in[7];
  const float* WcF  = (const float*)d_in[8];
  const float* WihB = (const float*)d_in[9];
  const float* WhhB = (const float*)d_in[10];
  const float* bB   = (const float*)d_in[11];
  const float* WcB  = (const float*)d_in[12];

  char* ws = (char*)d_ws;
  size_t off = 0;
  u16* Xbf   = (u16*)(ws + off); off += (size_t)TB * IDIM * 2;        // 32 MB
  u16* Wcat  = (u16*)(ws + off); off += (size_t)2 * H3 * IDIM * 2;    // 3 MB
  u16* WhhBf = (u16*)(ws + off); off += (size_t)2 * H2 * HDIM * 2;    // 2 MB
  float* cWS = (float*)(ws + off); off += (size_t)2 * BATCH * HDIM * 4;
  float* Pre = (float*)(ws + off); off += (size_t)2 * TB * H3 * 4;    // 384 MB

  auto cvt = [&](const float* s, u16* dd, int n) {
    cvt_bf16_kernel<<<dim3((n + 255) / 256), dim3(256), 0, stream>>>(s, dd, n);
  };
  cvt(X,    Xbf, TB * IDIM);
  cvt(WihF, Wcat,                                     H2 * IDIM);
  cvt(WcF,  Wcat + (size_t)H2 * IDIM,                 HDIM * IDIM);
  cvt(WihB, Wcat + (size_t)H3 * IDIM,                 H2 * IDIM);
  cvt(WcB,  Wcat + (size_t)H3 * IDIM + (size_t)H2 * IDIM, HDIM * IDIM);
  cvt(WhhF, WhhBf,                     H2 * HDIM);
  cvt(WhhB, WhhBf + (size_t)H2 * HDIM, H2 * HDIM);

  pregemm_kernel<<<dim3(TB / 128, H3 / 64, 2), dim3(256), 0, stream>>>(
      Xbf, Wcat, bF, bB, Pre);

  regu_recurrent_kernel<<<dim3(2), dim3(512), 0, stream>>>(
      X, WhhBf, Pre, h0F, c0F, h0B, c0B, cWS, (float*)d_out);
}
